// dilate_block_31782757990656
// MI455X (gfx1250) — matrix operand order repaired and checked by reading, not compiled, not run on hardware
//
#include <hip/hip_runtime.h>

typedef unsigned short u16;
typedef __attribute__((ext_vector_type(16))) __bf16 v16bf;
typedef __attribute__((ext_vector_type(8)))  __bf16 v8bf;
typedef __attribute__((ext_vector_type(8)))  float  v8f;

union AU { v16bf v; v8bf h[2]; };

__device__ __forceinline__ u16 f2bf(float f) {
  unsigned int u = __float_as_uint(f);
  unsigned int r = (u + 0x7fffu + ((u >> 16) & 1u)) >> 16;   // round-to-nearest-even
  return (u16)r;
}

// ---- pack f32 weights [taps][Cin][64] into WMMA B-fragment lane layout (bf16) ----
// blob linear index: (((k*(Cin/32)+c)*4 + t)*32 + lane)*16 + e
// element maps to  W[k][c*32 + 8*half + (e&7) + 16*(e>>3)][t*16 + (lane&15)]
__global__ void pack_w(const float* __restrict__ w, u16* __restrict__ dst,
                       int taps, int Cin) {
  int i = blockIdx.x * blockDim.x + threadIdx.x;
  int total = taps * Cin * 64;
  if (i >= total) return;
  int e    = i & 15;
  int lane = (i >> 4) & 31;
  int t    = (i >> 9) & 3;
  int rest = i >> 11;                 // k*(Cin/32)+c
  int nch  = Cin >> 5;
  int k = rest / nch, c = rest % nch;
  int half = lane >> 4, n = lane & 15;
  int cin  = (c << 5) + (half << 3) + (e & 7) + ((e >> 3) << 4);
  int cout = (t << 4) + n;
  dst[i] = f2bf(w[(k * Cin + cin) * 64 + cout]);
}

// ---- convert input features f32 -> bf16, zero the pad row (row N) ----
__global__ void cvt_feats(const float* __restrict__ src, u16* __restrict__ dst, int N) {
  int i = blockIdx.x * blockDim.x + threadIdx.x;
  int total = (N + 1) * 32;
  if (i >= total) return;
  int r = i >> 5;
  dst[i] = (r == N) ? (u16)0 : f2bf(src[i]);
}

__global__ void zero_stats(float* s) { if (threadIdx.x < 128) s[threadIdx.x] = 0.0f; }

// per-channel sum / sumsq over rows (C = 64)
__global__ void bn_stats(const float* __restrict__ x, int rows, float* __restrict__ stats) {
  int tid = blockIdx.x * blockDim.x + threadIdx.x;
  int c = tid & 63;
  int r0 = tid >> 6;
  int rstep = (gridDim.x * blockDim.x) >> 6;
  float s = 0.0f, q = 0.0f;
  for (int r = r0; r < rows; r += rstep) {
    float v = x[(size_t)r * 64 + c];
    s += v; q += v * v;
  }
  atomicAdd(&stats[c], s);
  atomicAdd(&stats[64 + c], q);
}

// BN(train stats)+LeakyReLU (optional), + residual add (optional),
// writes f32 (optional) and/or bf16 with given row stride (optional, pads row==rows with 0)
__global__ void bn_apply(const float* __restrict__ x, const float* __restrict__ stats,
                         int rows, const float* __restrict__ addsrc,
                         float* __restrict__ out_f32,
                         u16* __restrict__ out_bf16, int bstride, int do_bn) {
  int i = blockIdx.x * blockDim.x + threadIdx.x;
  int total = (rows + 1) * 64;
  if (i >= total) return;
  int r = i >> 6, c = i & 63;
  if (r == rows) {                      // pad row for gather sentinel
    if (out_bf16) out_bf16[(size_t)r * bstride + c] = 0;
    return;
  }
  float v = x[i];
  if (do_bn) {
    float inv = 1.0f / (float)rows;
    float m  = stats[c] * inv;
    float vr = stats[64 + c] * inv - m * m;
    v = (v - m) * rsqrtf(vr + 1e-5f);
    v = v > 0.0f ? v : 0.01f * v;       // LeakyReLU
  }
  if (addsrc) v += addsrc[i];
  if (out_f32) out_f32[i] = v;
  if (out_bf16) out_bf16[(size_t)r * bstride + c] = f2bf(v);
}

// ---- gather + WMMA GEMM sparse conv: out[n,:] = sum_k src[nbr[k][n]] @ W[k] ----
// One wave -> 32 output rows x 64 output channels (8 WMMA C-tiles, f32 accum).
// B fragments are reused across the two 16-row sub-tiles: 1.0 b128 load per WMMA.
__global__ __launch_bounds__(256) void subm_conv(
    const u16* __restrict__ src, int src_stride, int src_rows,
    const u16* __restrict__ wp, const int* __restrict__ nbr,
    float* __restrict__ out, int out_rows, int Cin, int taps) {
  const int wave  = blockIdx.x * (blockDim.x >> 5) + (threadIdx.x >> 5);
  const int lane  = threadIdx.x & 31;
  const int tile0 = wave << 5;          // 32 rows per wave
  if (tile0 >= out_rows) return;        // wave-uniform exit (EXEC stays full below)
  const int half = lane >> 4;
  const int m    = lane & 15;
  const int row0 = tile0 + m;
  const int row1 = tile0 + 16 + m;
  const int nch  = Cin >> 5;

  v8f p0 = {}, p1 = {}, p2 = {}, p3 = {};   // rows [tile0, tile0+16)
  v8f q0 = {}, q1 = {}, q2 = {}, q3 = {};   // rows [tile0+16, tile0+32)
  for (int k = 0; k < taps; ++k) {
    int g0 = src_rows, g1 = src_rows;   // sentinel -> zero pad row
    if (row0 < out_rows) g0 = nbr ? nbr[k * out_rows + row0] : row0;
    if (row1 < out_rows) g1 = nbr ? nbr[k * out_rows + row1] : row1;
    const u16* arow0 = src + (size_t)g0 * src_stride;
    const u16* arow1 = src + (size_t)g1 * src_stride;
    const u16* wk    = wp + (size_t)k * nch * 2048;
    for (int c = 0; c < nch; ++c) {
      const int co = (c << 5) + (half << 3);
      AU a0, a1;                        // ISA 16-bit A 16x32 lane layout
      a0.h[0] = *(const v8bf*)(arow0 + co);
      a0.h[1] = *(const v8bf*)(arow0 + co + 16);
      a1.h[0] = *(const v8bf*)(arow1 + co);
      a1.h[1] = *(const v8bf*)(arow1 + co + 16);
      const u16* bb = wk + c * 2048 + lane * 16;
      v16bf b0 = *(const v16bf*)(bb);
      v16bf b1 = *(const v16bf*)(bb + 512);
      v16bf b2 = *(const v16bf*)(bb + 1024);
      v16bf b3 = *(const v16bf*)(bb + 1536);
      p0 = __builtin_amdgcn_wmma_f32_16x16x32_bf16(false, a0.v, false, b0, (short)0, p0, false, false);
      q0 = __builtin_amdgcn_wmma_f32_16x16x32_bf16(false, a1.v, false, b0, (short)0, q0, false, false);
      p1 = __builtin_amdgcn_wmma_f32_16x16x32_bf16(false, a0.v, false, b1, (short)0, p1, false, false);
      q1 = __builtin_amdgcn_wmma_f32_16x16x32_bf16(false, a1.v, false, b1, (short)0, q1, false, false);
      p2 = __builtin_amdgcn_wmma_f32_16x16x32_bf16(false, a0.v, false, b2, (short)0, p2, false, false);
      q2 = __builtin_amdgcn_wmma_f32_16x16x32_bf16(false, a1.v, false, b2, (short)0, q2, false, false);
      p3 = __builtin_amdgcn_wmma_f32_16x16x32_bf16(false, a0.v, false, b3, (short)0, p3, false, false);
      q3 = __builtin_amdgcn_wmma_f32_16x16x32_bf16(false, a1.v, false, b3, (short)0, q3, false, false);
    }
  }
  for (int r = 0; r < 8; ++r) {         // C-tile layout: VGPR r -> M=r (+8 for upper half)
    int mr0 = tile0 + r + (half << 3);
    if (mr0 < out_rows) {
      float* o = out + (size_t)mr0 * 64 + m;
      o[0]  = p0[r];
      o[16] = p1[r];
      o[32] = p2[r];
      o[48] = p3[r];
    }
    int mr1 = mr0 + 16;
    if (mr1 < out_rows) {
      float* o = out + (size_t)mr1 * 64 + m;
      o[0]  = q0[r];
      o[16] = q1[r];
      o[32] = q2[r];
      o[48] = q3[r];
    }
  }
}

// ---------------- host orchestration ----------------
static void run_conv(const u16* src, int stride, int srows, const u16* wp, const int* nbr,
                     float* out, int orows, int Cin, int taps, hipStream_t s) {
  int blocks = (orows + 255) / 256;     // 8 waves/block * 32 rows/wave
  subm_conv<<<blocks, 256, 0, s>>>(src, stride, srows, wp, nbr, out, orows, Cin, taps);
}
static void run_bn(float* x, int rows, float* stats, const float* add, float* of32,
                   u16* obf, int bstride, int do_bn, hipStream_t s) {
  if (do_bn) {
    zero_stats<<<1, 128, 0, s>>>(stats);
    bn_stats<<<160, 256, 0, s>>>(x, rows, stats);
  }
  int total = (rows + 1) * 64;
  bn_apply<<<(total + 255) / 256, 256, 0, s>>>(x, stats, rows, add, of32, obf, bstride, do_bn);
}

extern "C" void kernel_launch(void* const* d_in, const int* in_sizes, int n_in,
                              void* d_out, int out_size, void* d_ws, size_t ws_size,
                              hipStream_t stream) {
  (void)n_in; (void)out_size; (void)ws_size;
  const float* feats = (const float*)d_in[0];
  const float* W[14];
  for (int i = 0; i < 14; ++i) W[i] = (const float*)d_in[1 + i];
  const int* nbrA1 = (const int*)d_in[15];
  const int* nbrB1 = (const int*)d_in[16];
  const int* pool  = (const int*)d_in[17];
  const int* nbrA2 = (const int*)d_in[18];
  const int* nbrB2 = (const int*)d_in[19];
  const int* nbrA3 = (const int*)d_in[20];
  const int* nbrB3 = (const int*)d_in[21];

  const int N = in_sizes[0] / 32;
  const int M = in_sizes[17] / 27;
  const int rmax = N > M ? N : M;

  // carve workspace (256B aligned)
  char* p = (char*)d_ws;
  auto carve = [&](size_t bytes) -> char* {
    char* r = p; p += (bytes + 255) & ~(size_t)255; return r;
  };
  static const int wtaps[14] = {9,9,9,9,27, 9,9,9,9, 9,9,9,9, 1};
  static const int wcin [14] = {32,64,32,64,64, 64,64,64,64, 64,64,64,64, 192};
  size_t woff[14], wtot = 0;
  for (int i = 0; i < 14; ++i) { woff[i] = wtot; wtot += (size_t)wtaps[i] * wcin[i] * 64; }
  u16*   wpack = (u16*)carve(wtot * 2);
  u16*   finb  = (u16*)carve((size_t)(N + 1) * 32 * 2);
  u16*   fb0   = (u16*)carve((size_t)(rmax + 1) * 64 * 2);
  u16*   fb1   = (u16*)carve((size_t)(rmax + 1) * 64 * 2);
  u16*   xcat  = (u16*)carve((size_t)(M + 1) * 192 * 2);
  float* cf    = (float*)carve((size_t)rmax * 64 * 4);
  float* sf    = (float*)carve((size_t)rmax * 64 * 4);
  float* stats = (float*)carve(128 * 4);

  // pack all weights into WMMA B layout (bf16)
  for (int i = 0; i < 14; ++i) {
    int total = wtaps[i] * wcin[i] * 64;
    pack_w<<<(total + 255) / 256, 256, 0, stream>>>(W[i], wpack + woff[i], wtaps[i], wcin[i]);
  }
  cvt_feats<<<((N + 1) * 32 + 255) / 256, 256, 0, stream>>>(feats, finb, N);

  // ---- Block 1 (N rows): s = bnA(conv nA w1) -> bnA(conv nB w1_2); r = bnA(conv nB w2) -> bnA(conv nA w2_2); x1 = r+s
  run_conv(finb, 32, N, wpack + woff[0], nbrA1, cf, N, 32, 9, stream);
  run_bn(cf, N, stats, nullptr, nullptr, fb0, 64, 1, stream);
  run_conv(fb0, 64, N, wpack + woff[1], nbrB1, cf, N, 64, 9, stream);
  run_bn(cf, N, stats, nullptr, sf, nullptr, 0, 1, stream);
  run_conv(finb, 32, N, wpack + woff[2], nbrB1, cf, N, 32, 9, stream);
  run_bn(cf, N, stats, nullptr, nullptr, fb0, 64, 1, stream);
  run_conv(fb0, 64, N, wpack + woff[3], nbrA1, cf, N, 64, 9, stream);
  run_bn(cf, N, stats, sf, nullptr, fb1, 64, 1, stream);           // x1 (bf16, N rows)

  // ---- Pool: strided 27-tap conv N -> M, no BN/act; write into xcat cols [0,64)
  run_conv(fb1, 64, N, wpack + woff[4], pool, cf, M, 64, 27, stream);
  run_bn(cf, M, stats, nullptr, nullptr, xcat, 192, 0, stream);    // x1p

  // ---- Block 2 (M rows, dilation-2 maps), input xcat cols [0,64)
  run_conv(xcat, 192, M, wpack + woff[5], nbrA2, cf, M, 64, 9, stream);
  run_bn(cf, M, stats, nullptr, nullptr, fb0, 64, 1, stream);
  run_conv(fb0, 64, M, wpack + woff[6], nbrB2, cf, M, 64, 9, stream);
  run_bn(cf, M, stats, nullptr, sf, nullptr, 0, 1, stream);
  run_conv(xcat, 192, M, wpack + woff[7], nbrB2, cf, M, 64, 9, stream);
  run_bn(cf, M, stats, nullptr, nullptr, fb0, 64, 1, stream);
  run_conv(fb0, 64, M, wpack + woff[8], nbrA2, cf, M, 64, 9, stream);
  run_bn(cf, M, stats, sf, nullptr, xcat + 64, 192, 1, stream);    // x2 -> cols [64,128)

  // ---- Block 3 (M rows, dilation-3 maps), input xcat cols [64,128)
  run_conv(xcat + 64, 192, M, wpack + woff[9], nbrA3, cf, M, 64, 9, stream);
  run_bn(cf, M, stats, nullptr, nullptr, fb0, 64, 1, stream);
  run_conv(fb0, 64, M, wpack + woff[10], nbrB3, cf, M, 64, 9, stream);
  run_bn(cf, M, stats, nullptr, sf, nullptr, 0, 1, stream);
  run_conv(xcat + 64, 192, M, wpack + woff[11], nbrB3, cf, M, 64, 9, stream);
  run_bn(cf, M, stats, nullptr, nullptr, fb0, 64, 1, stream);
  run_conv(fb0, 64, M, wpack + woff[12], nbrA3, cf, M, 64, 9, stream);
  run_bn(cf, M, stats, sf, nullptr, xcat + 128, 192, 1, stream);   // x3 -> cols [128,192)

  // ---- Final 1x1: concat(192) @ down_w -> d_out [M,64] (identity map, 1 tap, WMMA)
  run_conv(xcat, 192, M, wpack + woff[13], nullptr, (float*)d_out, M, 192, 1, stream);
}
